// RobertaCRFForTokenClassification_1030792151081
// MI455X (gfx1250) — matrix operand order repaired and checked by reading, not compiled, not run on hardware
//
#include <hip/hip_runtime.h>
#include <cstdint>

// ---------------------------------------------------------------------------
// Problem constants (match reference)
// ---------------------------------------------------------------------------
#define B_  4
#define S_  1024
#define H_  1024
#define NH_ 16
#define D_  64
#define L_  9
#define M_  (B_ * S_)          // 4096 token rows

typedef __attribute__((ext_vector_type(16))) _Float16 v16h;
typedef __attribute__((ext_vector_type(8)))  float    v8f;

#define WMMA_F16(a, b, c) \
  __builtin_amdgcn_wmma_f32_16x16x32_f16(false, (a), false, (b), (short)0, (c), false, false)

// Load 16 contiguous halves (two b128) -> fragment elems 0..15
__device__ __forceinline__ v16h ld_frag16(const _Float16* p) {
  v16h f;
  uint4* fp = reinterpret_cast<uint4*>(&f);
  fp[0] = *reinterpret_cast<const uint4*>(p);
  fp[1] = *reinterpret_cast<const uint4*>(p + 8);
  return f;
}

// A-layout fragment: elems 0..7 at p, elems 8..15 at p+16 (gap pattern)
__device__ __forceinline__ v16h ld_a_frag(const _Float16* p) {
  v16h f;
  uint4* fp = reinterpret_cast<uint4*>(&f);
  fp[0] = *reinterpret_cast<const uint4*>(p);
  fp[1] = *reinterpret_cast<const uint4*>(p + 16);
  return f;
}

// ---------------------------------------------------------------------------
// 1) fp32 -> fp16 elementwise conversion
// ---------------------------------------------------------------------------
__global__ void cvt_f32_to_f16(const float* __restrict__ src,
                               _Float16* __restrict__ dst, int n) {
  int i = blockIdx.x * blockDim.x + threadIdx.x;
  if (i < n) dst[i] = (_Float16)src[i];
}

// ---------------------------------------------------------------------------
// 2/4) WMMA GEMM: out = A[M,K](f16) @ W[K,N](f16) + bias (+residual)
//      Block: 256 thr (8 waves) -> 256x64 tile; wave -> 32x64.
//      Double-buffered LDS W tile (transposed [n][k]); software pipeline:
//      load next tile to regs -> WMMA current -> scatter regs -> 1 barrier.
// ---------------------------------------------------------------------------
__global__ __launch_bounds__(256)
void gemm_wmma(const _Float16* __restrict__ A, const _Float16* __restrict__ W,
               const float* __restrict__ bias, const float* __restrict__ residual,
               _Float16* __restrict__ outH, float* __restrict__ outF,
               int M, int N, int K) {
  // rows padded to 40 halves (80B): 16-lane b128 reads hit distinct banks
  __shared__ alignas(16) _Float16 WldsT[2][64][40];

  const int tid  = threadIdx.x;
  const int wave = tid >> 5;
  const int lane = tid & 31;
  const int hb   = lane >> 4;     // half-wave bit
  const int l16  = lane & 15;
  const int m0   = blockIdx.x * 256 + wave * 32;
  const int n0   = blockIdx.y * 64;

  const v8f zacc = {0.f, 0.f, 0.f, 0.f, 0.f, 0.f, 0.f, 0.f};
  v8f acc[2][4];
#pragma unroll
  for (int mt = 0; mt < 2; ++mt)
#pragma unroll
    for (int t = 0; t < 4; ++t) acc[mt][t] = zacc;

  const int sr = tid >> 3;          // staging k-row 0..31
  const int sc = (tid & 7) * 8;     // staging n-col 0..56 (8 halves = 16B)

  const _Float16* arow0 = A + (size_t)(m0 + l16) * K + hb * 8;
  const _Float16* arow1 = A + (size_t)(m0 + 16 + l16) * K + hb * 8;

  // ---- prologue: stage tile 0, preload A fragments for iter 0 ----
  {
    const uint4 w4 = *reinterpret_cast<const uint4*>(&W[(size_t)sr * N + n0 + sc]);
    const _Float16* wh = reinterpret_cast<const _Float16*>(&w4);
#pragma unroll
    for (int j = 0; j < 8; ++j) WldsT[0][sc + j][sr] = wh[j];
  }
  v16h a0 = ld_a_frag(arow0);
  v16h a1 = ld_a_frag(arow1);
  __syncthreads();

  const int nIter = K >> 5;
  for (int i = 0; i < nIter; ++i) {
    const int cur  = i & 1;
    const bool more = (i + 1 < nIter);
    const int k1 = (i + 1) << 5;

    // issue next tile's global loads early (consumed after the WMMAs)
    uint4 wn;
    v16h na0, na1;
    if (more) {
      wn  = *reinterpret_cast<const uint4*>(&W[(size_t)(k1 + sr) * N + n0 + sc]);
      na0 = ld_a_frag(arow0 + k1);
      na1 = ld_a_frag(arow1 + k1);
      if (i + 2 < nIter)
        __builtin_prefetch(&W[(size_t)(k1 + 32 + sr) * N + n0 + sc], 0, 3);
    }

    // preload ALL B fragments (distinct regs -> staggered dscnt waits)
    v16h bf[4];
#pragma unroll
    for (int t = 0; t < 4; ++t)
      bf[t] = ld_a_frag(&WldsT[cur][t * 16 + l16][hb * 8]);

#pragma unroll
    for (int t = 0; t < 4; ++t) {
      acc[0][t] = WMMA_F16(a0, bf[t], acc[0][t]);
      acc[1][t] = WMMA_F16(a1, bf[t], acc[1][t]);
    }

    if (more) {
      const _Float16* wh = reinterpret_cast<const _Float16*>(&wn);
#pragma unroll
      for (int j = 0; j < 8; ++j) WldsT[cur ^ 1][sc + j][sr] = wh[j];
      a0 = na0;
      a1 = na1;
    }
    __syncthreads();   // single barrier per k-step
  }

  // ---- epilogue: bias (+residual), store f16 and/or f32 ----
#pragma unroll
  for (int mt = 0; mt < 2; ++mt) {
#pragma unroll
    for (int t = 0; t < 4; ++t) {
      const int n  = n0 + t * 16 + l16;
      const float bn = bias ? bias[n] : 0.f;
#pragma unroll
      for (int r = 0; r < 8; ++r) {
        const int m = m0 + mt * 16 + r + hb * 8;
        float v = acc[mt][t][r] + bn;
        if (residual) v += residual[(size_t)m * N + n];
        if (outF) outF[(size_t)m * N + n] = v;
        if (outH) outH[(size_t)m * N + n] = (_Float16)v;
      }
    }
  }
}

// ---------------------------------------------------------------------------
// 3) Flash attention per (b,h). Block = 8 waves sharing one (b,h);
//    each wave owns a 16-query tile; 32-key K/V tiles double-buffered in LDS.
//    K staged [key][d] (score B frags contiguous in d);
//    V staged transposed [d][key] (ctx B frags contiguous in key).
// ---------------------------------------------------------------------------
__global__ __launch_bounds__(256)
void attn_wmma(const _Float16* __restrict__ Q, const _Float16* __restrict__ K,
               const _Float16* __restrict__ V, _Float16* __restrict__ Ctx) {
  __shared__ alignas(16) _Float16 Klds[2][32][72];     // pad: 144B rows
  __shared__ alignas(16) _Float16 VldsT[2][64][40];    // pad: 80B rows
  __shared__ alignas(16) _Float16 Plds[8][16][40];     // per-wave P tile

  const int tid  = threadIdx.x;
  const int wave = tid >> 5;
  const int lane = tid & 31;
  const int hb   = lane >> 4;
  const int l16  = lane & 15;

  const int bh = blockIdx.x >> 3;            // 0..63
  const int qg = blockIdx.x & 7;             // query-tile group
  const int b  = bh >> 4;
  const int h  = bh & (NH_ - 1);
  const int q0 = (qg * 8 + wave) * 16;       // this wave's query-tile base

  const size_t baseBS = (size_t)b * S_ * H_;
  const size_t hoff   = (size_t)h * D_;
  const _Float16* Kbase = K + baseBS + hoff;
  const _Float16* Vbase = V + baseBS + hoff;

  // Q A-fragments for d = 0..31 and d = 32..63
  const _Float16* qrow = Q + baseBS + (size_t)(q0 + l16) * H_ + hoff + hb * 8;
  const v16h qa0 = ld_a_frag(qrow);
  const v16h qa1 = ld_a_frag(qrow + 32);

  const v8f zacc = {0.f, 0.f, 0.f, 0.f, 0.f, 0.f, 0.f, 0.f};
  v8f acc[4];
#pragma unroll
  for (int t = 0; t < 4; ++t) acc[t] = zacc;
  float mrow[8], lrow[8];
#pragma unroll
  for (int r = 0; r < 8; ++r) { mrow[r] = -1e30f; lrow[r] = 0.f; }

  const int sr = tid >> 3;          // staging key-row 0..31
  const int sc = (tid & 7) * 8;     // staging d-col 0..56

  // ---- prologue: stage K/V tile 0 ----
  {
    *(uint4*)&Klds[0][sr][sc] =
        *(const uint4*)&Kbase[(size_t)sr * H_ + sc];
    const uint4 v4 = *(const uint4*)&Vbase[(size_t)sr * H_ + sc];
    const _Float16* vh = reinterpret_cast<const _Float16*>(&v4);
#pragma unroll
    for (int j = 0; j < 8; ++j) VldsT[0][sc + j][sr] = vh[j];
  }
  __syncthreads();

  const int nIter = S_ >> 5;   // 32
  for (int it = 0; it < nIter; ++it) {
    const int cur  = it & 1;
    const int kb   = it << 5;
    const bool more = (it + 1 < nIter);

    // issue next tile's global loads early
    uint4 kn, vn;
    if (more) {
      kn = *(const uint4*)&Kbase[(size_t)(kb + 32 + sr) * H_ + sc];
      vn = *(const uint4*)&Vbase[(size_t)(kb + 32 + sr) * H_ + sc];
      if (it + 2 < nIter) {
        __builtin_prefetch(&Kbase[(size_t)(kb + 64 + sr) * H_ + sc], 0, 3);
        __builtin_prefetch(&Vbase[(size_t)(kb + 64 + sr) * H_ + sc], 0, 3);
      }
    }

    // preload all K-score and V fragments from current buffers
    v16h kf0[2], kf1[2], vf[4];
#pragma unroll
    for (int j = 0; j < 2; ++j) {
      const _Float16* krow = &Klds[cur][j * 16 + l16][0];
      kf0[j] = ld_a_frag(krow + hb * 8);
      kf1[j] = ld_a_frag(krow + 32 + hb * 8);
    }
#pragma unroll
    for (int t = 0; t < 4; ++t)
      vf[t] = ld_a_frag(&VldsT[cur][t * 16 + l16][hb * 8]);

    // ---- scores for two 16-key subtiles (contract over D=64) ----
    v8f s[2];
#pragma unroll
    for (int j = 0; j < 2; ++j) {
      s[j] = WMMA_F16(qa0, kf0[j], zacc);
      s[j] = WMMA_F16(qa1, kf1[j], s[j]);
    }

    // ---- bias + new row max ----
    float rmax[8];
#pragma unroll
    for (int r = 0; r < 8; ++r) rmax[r] = mrow[r];
#pragma unroll
    for (int j = 0; j < 2; ++j) {
#pragma unroll
      for (int r = 0; r < 8; ++r) {
        const float qi = (float)(q0 + r + hb * 8);
        const float ki = (float)(kb + j * 16 + l16);
        const float dist = fabsf(qi - ki);
        const float pb = __expf(-0.1f * fminf(dist, 5.f));
        const float v = (s[j][r] + pb) * 0.125f - 0.1f * dist;
        s[j][r] = v;
        rmax[r] = fmaxf(rmax[r], v);
      }
    }
#pragma unroll
    for (int r = 0; r < 8; ++r) {   // half-wave max reduce
      float v = rmax[r];
      v = fmaxf(v, __shfl_xor(v, 1, 16));
      v = fmaxf(v, __shfl_xor(v, 2, 16));
      v = fmaxf(v, __shfl_xor(v, 4, 16));
      v = fmaxf(v, __shfl_xor(v, 8, 16));
      rmax[r] = v;                  // m_new
    }

    // ---- rescale old state, exponentiate, stash P to LDS ----
#pragma unroll
    for (int r = 0; r < 8; ++r) {
      const float scale = __expf(mrow[r] - rmax[r]);
      mrow[r] = rmax[r];
      lrow[r] *= scale;
#pragma unroll
      for (int t = 0; t < 4; ++t) acc[t][r] *= scale;
      float ps = 0.f;
#pragma unroll
      for (int j = 0; j < 2; ++j) {
        const float p = __expf(s[j][r] - rmax[r]);
        ps += p;
        Plds[wave][r + hb * 8][j * 16 + l16] = (_Float16)p;
      }
      ps += __shfl_xor(ps, 1, 16);
      ps += __shfl_xor(ps, 2, 16);
      ps += __shfl_xor(ps, 4, 16);
      ps += __shfl_xor(ps, 8, 16);
      lrow[r] += ps;
    }

    // ---- ctx += P(16x32) @ V(32x64) ----
    const v16h pa = ld_a_frag(&Plds[wave][l16][hb * 8]);
#pragma unroll
    for (int t = 0; t < 4; ++t)
      acc[t] = WMMA_F16(pa, vf[t], acc[t]);

    // scatter next tile into the other buffers
    if (more) {
      *(uint4*)&Klds[cur ^ 1][sr][sc] = kn;
      const _Float16* vh = reinterpret_cast<const _Float16*>(&vn);
#pragma unroll
      for (int j = 0; j < 8; ++j) VldsT[cur ^ 1][sc + j][sr] = vh[j];
    }
    __syncthreads();   // single barrier per key-step
  }

  // normalize and write context (f16, [B,S,H] with head offset)
#pragma unroll
  for (int t = 0; t < 4; ++t) {
#pragma unroll
    for (int r = 0; r < 8; ++r) {
      const float o = acc[t][r] / lrow[r];
      Ctx[baseBS + (size_t)(q0 + r + hb * 8) * H_ + hoff + t * 16 + l16] =
          (_Float16)o;
    }
  }
}

// ---------------------------------------------------------------------------
// 5) LayerNorm + classifier (span_logits), wave-per-token, Ws/gamma/beta in LDS
// ---------------------------------------------------------------------------
__global__ __launch_bounds__(256)
void ln_classify(const float* __restrict__ Y, const float* __restrict__ g,
                 const float* __restrict__ bln, const float* __restrict__ Ws,
                 const float* __restrict__ bs, float* __restrict__ Span) {
  __shared__ float WsLds[H_ * L_];
  __shared__ float gLds[H_];
  __shared__ float bLds[H_];
  __shared__ float bsLds[L_];

  for (int i = threadIdx.x; i < H_ * L_; i += 256) WsLds[i] = Ws[i];
  for (int i = threadIdx.x; i < H_; i += 256) { gLds[i] = g[i]; bLds[i] = bln[i]; }
  if (threadIdx.x < L_) bsLds[threadIdx.x] = bs[threadIdx.x];
  __syncthreads();

  const int wave = threadIdx.x >> 5;
  const int lane = threadIdx.x & 31;
  const int tok  = blockIdx.x * 8 + wave;
  const float* y = Y + (size_t)tok * H_;

  float x[32];
  float sum = 0.f, sq = 0.f;
#pragma unroll
  for (int i = 0; i < 32; ++i) {
    const float v = y[i * 32 + lane];
    x[i] = v; sum += v; sq += v * v;
  }
#pragma unroll
  for (int m = 1; m < 32; m <<= 1) {
    sum += __shfl_xor(sum, m, 32);
    sq  += __shfl_xor(sq,  m, 32);
  }
  const float mu  = sum * (1.f / H_);
  const float var = sq * (1.f / H_) - mu * mu;          // biased var (jnp.var)
  const float inv = rsqrtf(var + 1e-5f);

  float lg[L_];
#pragma unroll
  for (int l = 0; l < L_; ++l) lg[l] = 0.f;
#pragma unroll
  for (int i = 0; i < 32; ++i) {
    const int hh = i * 32 + lane;
    const float yn = (x[i] - mu) * inv * gLds[hh] + bLds[hh];
#pragma unroll
    for (int l = 0; l < L_; ++l) lg[l] += yn * WsLds[hh * L_ + l];
  }
#pragma unroll
  for (int l = 0; l < L_; ++l) {
#pragma unroll
    for (int m = 1; m < 32; m <<= 1) lg[l] += __shfl_xor(lg[l], m, 32);
  }
  if (lane == 0) {
#pragma unroll
    for (int l = 0; l < L_; ++l) Span[(size_t)tok * L_ + l] = lg[l] + bsLds[l];
  }
}

// ---------------------------------------------------------------------------
// 6) Entity bias: reads pristine pre-bias logits, writes final output
// ---------------------------------------------------------------------------
__global__ void entity_bias(const float* __restrict__ Span,
                            const float* __restrict__ eb,
                            float* __restrict__ Out) {
  const int t = blockIdx.x * blockDim.x + threadIdx.x;
  if (t >= M_) return;
  const int j = t & (S_ - 1);
  const float* row = Span + (size_t)t * L_;
  float v[L_];
#pragma unroll
  for (int l = 0; l < L_; ++l) v[l] = row[l];
  if (j > 0) {
    const float* pr = row - L_;           // previous token, pre-bias
    float best = pr[0]; int bi = 0;
#pragma unroll
    for (int l = 1; l < L_; ++l) {
      const float p = pr[l];
      if (p > best) { best = p; bi = l; } // first-max = jnp.argmax
    }
    if (bi == 1) v[2] += 2.f * eb[2];     // B-PERSON -> boost I-PERSON
  }
#pragma unroll
  for (int l = 0; l < L_; ++l) Out[(size_t)t * L_ + l] = v[l];
}

// ---------------------------------------------------------------------------
// Host-side orchestration
// ---------------------------------------------------------------------------
extern "C" void kernel_launch(void* const* d_in, const int* in_sizes, int n_in,
                              void* d_out, int out_size, void* d_ws, size_t ws_size,
                              hipStream_t stream) {
  (void)in_sizes; (void)n_in; (void)out_size; (void)ws_size;

  const float* X    = (const float*)d_in[0];
  const float* Wq   = (const float*)d_in[1];
  const float* bq   = (const float*)d_in[2];
  const float* Wk   = (const float*)d_in[3];
  const float* bk   = (const float*)d_in[4];
  const float* Wv   = (const float*)d_in[5];
  const float* bv   = (const float*)d_in[6];
  const float* Wo   = (const float*)d_in[7];
  const float* bo   = (const float*)d_in[8];
  const float* ln_g = (const float*)d_in[9];
  const float* ln_b = (const float*)d_in[10];
  const float* Ws   = (const float*)d_in[11];
  const float* bs   = (const float*)d_in[12];
  const float* eb   = (const float*)d_in[13];
  // d_in[14] = rel_pos_emb: unused in forward math (matches reference)

  char* ws = (char*)d_ws;
  const size_t MiB = 1024 * 1024;
  _Float16* Xh   = (_Float16*)(ws + 0 * MiB);    // 8 MiB
  _Float16* Wqh  = (_Float16*)(ws + 8 * MiB);    // 2 MiB each
  _Float16* Wkh  = (_Float16*)(ws + 10 * MiB);
  _Float16* Wvh  = (_Float16*)(ws + 12 * MiB);
  _Float16* Woh  = (_Float16*)(ws + 14 * MiB);
  _Float16* Qh   = (_Float16*)(ws + 16 * MiB);   // 8 MiB each
  _Float16* Kh   = (_Float16*)(ws + 24 * MiB);
  _Float16* Vh   = (_Float16*)(ws + 32 * MiB);
  _Float16* Ctxh = (_Float16*)(ws + 40 * MiB);
  float*    Yf   = (float*)   (ws + 48 * MiB);   // 16 MiB
  float*    Span = (float*)   (ws + 64 * MiB);   // 144 KiB

  const int nX = M_ * H_;        // 4,194,304
  const int nW = H_ * H_;        // 1,048,576
  cvt_f32_to_f16<<<(nX + 255) / 256, 256, 0, stream>>>(X,  Xh,  nX);
  cvt_f32_to_f16<<<(nW + 255) / 256, 256, 0, stream>>>(Wq, Wqh, nW);
  cvt_f32_to_f16<<<(nW + 255) / 256, 256, 0, stream>>>(Wk, Wkh, nW);
  cvt_f32_to_f16<<<(nW + 255) / 256, 256, 0, stream>>>(Wv, Wvh, nW);
  cvt_f32_to_f16<<<(nW + 255) / 256, 256, 0, stream>>>(Wo, Woh, nW);

  dim3 gg(M_ / 256, H_ / 64);   // 16 x 16 blocks, 256x64 tile per block
  gemm_wmma<<<gg, 256, 0, stream>>>(Xh, Wqh, bq, nullptr, Qh, nullptr, M_, H_, H_);
  gemm_wmma<<<gg, 256, 0, stream>>>(Xh, Wkh, bk, nullptr, Kh, nullptr, M_, H_, H_);
  gemm_wmma<<<gg, 256, 0, stream>>>(Xh, Wvh, bv, nullptr, Vh, nullptr, M_, H_, H_);

  attn_wmma<<<B_ * NH_ * 8, 256, 0, stream>>>(Qh, Kh, Vh, Ctxh);

  // y = x + ctx @ Wo + bo  (f32 output with residual)
  gemm_wmma<<<gg, 256, 0, stream>>>(Ctxh, Woh, bo, X, nullptr, Yf, M_, H_, H_);

  ln_classify<<<M_ / 8, 256, 0, stream>>>(Yf, ln_g, ln_b, Ws, bs, Span);

  entity_bias<<<(M_ + 255) / 256, 256, 0, stream>>>(Span, eb, (float*)d_out);
}
